// MedformerLayer_23579370455566
// MI455X (gfx1250) — matrix operand order repaired and checked by reading, not compiled, not run on hardware
//
#include <hip/hip_runtime.h>
#include <hip/hip_bf16.h>

// ---------------------------------------------------------------------------
// Medformer-style layer for MI455X (gfx1250, wave32, WMMA).
// All GEMMs run through v_wmma_f32_16x16x32_bf16 (f32 accum), staged via LDS.
// ---------------------------------------------------------------------------

#define NBq  4
#define Bsz  16
#define Lseq 512
#define Dm   512
#define Hh   8
#define Dh   64
#define DCc  512
#define DFFc 2048

typedef __attribute__((ext_vector_type(16))) __bf16 v16bf;
typedef __attribute__((ext_vector_type(8)))  __bf16 v8bf;
typedef __attribute__((ext_vector_type(8)))  float  v8f;
typedef __attribute__((ext_vector_type(4)))  float  v4f;

union FragBF { v16bf v; v8bf h[2]; };
struct __align__(8) bf16x4 { __bf16 a, b, c, d; };

// ---------------------------------------------------------------------------
// Generic batched GEMM:  C[z] = act( alpha * A[z] x B[z] + bias )
//   A: [M,K] rows (lda), B: bLayout==0 -> [K,N] rows (ldb)
//                        bLayout==1 -> [N,K] rows (ldb)  (e.g. K-matrix for QK^T)
//   batch offset for X = (z/bh)*sX1 + (z%bh)*sX2   (covers [b,l,h,e] layouts)
// Tile: 128x128 per workgroup, 8 waves (4Mx2N), wave tile 32x64, K-step 32.
// ---------------------------------------------------------------------------
__global__ __launch_bounds__(256)
void wmma_gemm_kernel(const float* __restrict__ A, const float* __restrict__ Bm,
                      const float* __restrict__ bias, float* __restrict__ C,
                      int M, int N, int K, int lda, int ldb, int ldc,
                      int bh, long long sA1, long long sA2,
                      long long sB1, long long sB2,
                      long long sC1, long long sC2,
                      int bLayout, int act, float alpha)
{
    __shared__ __align__(16) __bf16 As[128][40];   // 128 rows x 32 K (+pad)
    __shared__ __align__(16) __bf16 Bs[128][40];   // 128 N   x 32 K (+pad), i.e. B^T

    const int tid   = threadIdx.x;
    const int lane  = tid & 31;
    const int wave  = tid >> 5;
    const int waveM = wave & 3;        // 4 waves along M
    const int waveN = wave >> 2;       // 2 waves along N
    const int kgrp  = lane >> 4;       // lane group 0/1
    const int l16   = lane & 15;

    const int z = blockIdx.z;
    const long long offA = (long long)(z / bh) * sA1 + (long long)(z % bh) * sA2;
    const long long offB = (long long)(z / bh) * sB1 + (long long)(z % bh) * sB2;
    const long long offC = (long long)(z / bh) * sC1 + (long long)(z % bh) * sC2;

    const int rowBase = blockIdx.y * 128;
    const int colBase = blockIdx.x * 128;

    const v8f vzero = {0.f,0.f,0.f,0.f,0.f,0.f,0.f,0.f};
    v8f acc[2][4];
#pragma unroll
    for (int i = 0; i < 2; ++i)
#pragma unroll
        for (int j = 0; j < 4; ++j) acc[i][j] = vzero;

    for (int k0 = 0; k0 < K; k0 += 32) {
        __syncthreads();
        // ---- stage A tile: 128 x 32 f32 -> bf16 (4 float4 per thread) ----
#pragma unroll
        for (int it = 0; it < 4; ++it) {
            int f  = tid + it * 256;      // 0..1023 float4s
            int r  = f >> 3;              // tile row
            int c4 = f & 7;               // float4 column
            int gr = rowBase + r;
            v4f val = {0.f, 0.f, 0.f, 0.f};
            if (gr < M)
                val = *(const v4f*)(A + offA + (long long)gr * lda + k0 + c4 * 4);
            bf16x4 hv; hv.a = (__bf16)val.x; hv.b = (__bf16)val.y;
                       hv.c = (__bf16)val.z; hv.d = (__bf16)val.w;
            *(bf16x4*)&As[r][c4 * 4] = hv;
        }
        // ---- stage B tile into B^T layout [N][K] ----
        if (bLayout == 0) {              // B global is [K, N] rows
#pragma unroll
            for (int it = 0; it < 4; ++it) {
                int f  = tid + it * 256;
                int kr = f >> 5;          // K row 0..31
                int n4 = f & 31;          // float4 along N
                int gn = colBase + n4 * 4;
                v4f val = {0.f, 0.f, 0.f, 0.f};
                if (gn + 3 < N) {
                    val = *(const v4f*)(Bm + offB + (long long)(k0 + kr) * ldb + gn);
                } else {
                    if (gn + 0 < N) val.x = Bm[offB + (long long)(k0 + kr) * ldb + gn + 0];
                    if (gn + 1 < N) val.y = Bm[offB + (long long)(k0 + kr) * ldb + gn + 1];
                    if (gn + 2 < N) val.z = Bm[offB + (long long)(k0 + kr) * ldb + gn + 2];
                }
                Bs[n4 * 4 + 0][kr] = (__bf16)val.x;
                Bs[n4 * 4 + 1][kr] = (__bf16)val.y;
                Bs[n4 * 4 + 2][kr] = (__bf16)val.z;
                Bs[n4 * 4 + 3][kr] = (__bf16)val.w;
            }
        } else {                          // B global is [N, K] rows (no transpose)
#pragma unroll
            for (int it = 0; it < 4; ++it) {
                int f  = tid + it * 256;
                int n  = f >> 3;
                int c4 = f & 7;
                int gn = colBase + n;
                v4f val = {0.f, 0.f, 0.f, 0.f};
                if (gn < N)
                    val = *(const v4f*)(Bm + offB + (long long)gn * ldb + k0 + c4 * 4);
                bf16x4 hv; hv.a = (__bf16)val.x; hv.b = (__bf16)val.y;
                           hv.c = (__bf16)val.z; hv.d = (__bf16)val.w;
                *(bf16x4*)&Bs[n][c4 * 4] = hv;
            }
        }
        __syncthreads();

        // ---- fragment loads (contiguous 16-K run per lane half) + 8 WMMAs per wave ----
        FragBF afr[2], bfr[4];
#pragma unroll
        for (int tm = 0; tm < 2; ++tm) {
            int r = waveM * 32 + tm * 16 + l16;          // A: contiguous 16-K run
            afr[tm].h[0] = *(const v8bf*)&As[r][kgrp * 16];
            afr[tm].h[1] = *(const v8bf*)&As[r][kgrp * 16 + 8];
        }
#pragma unroll
        for (int tn = 0; tn < 4; ++tn) {
            int r = waveN * 64 + tn * 16 + l16;          // B: contiguous 16-K run
            bfr[tn].h[0] = *(const v8bf*)&Bs[r][kgrp * 16];
            bfr[tn].h[1] = *(const v8bf*)&Bs[r][kgrp * 16 + 8];
        }
#pragma unroll
        for (int tm = 0; tm < 2; ++tm)
#pragma unroll
            for (int tn = 0; tn < 4; ++tn)
                acc[tm][tn] = __builtin_amdgcn_wmma_f32_16x16x32_bf16(
                    false, afr[tm].v, false, bfr[tn].v,
                    (short)0, acc[tm][tn], false, false);
    }

    // ---- epilogue: alpha, bias, activation ----
#pragma unroll
    for (int tm = 0; tm < 2; ++tm)
#pragma unroll
        for (int tn = 0; tn < 4; ++tn) {
            int col = colBase + waveN * 64 + tn * 16 + l16;
            if (col >= N) continue;
            float bv = bias ? bias[col] : 0.f;
#pragma unroll
            for (int r = 0; r < 8; ++r) {
                int row = rowBase + waveM * 32 + tm * 16 + r + kgrp * 8;
                if (row >= M) continue;
                float v = acc[tm][tn][r] * alpha + bv;
                if (act == 1)      v = fmaxf(v, 0.f);
                else if (act == 2) v = 0.5f * v * (1.f + erff(v * 0.70710678118f));
                C[offC + (long long)row * ldc + col] = v;
            }
        }
}

// ---------------------------------------------------------------------------
// Wave-wide reductions (wave32)
// ---------------------------------------------------------------------------
__device__ inline float wave_max(float v) {
#pragma unroll
    for (int o = 16; o; o >>= 1) v = fmaxf(v, __shfl_xor(v, o, 32));
    return v;
}
__device__ inline float wave_sum(float v) {
#pragma unroll
    for (int o = 16; o; o >>= 1) v += __shfl_xor(v, o, 32);
    return v;
}

// softmax over last dim, one wave per row
__global__ void softmax_rows_kernel(float* __restrict__ p, int rows, int cols) {
    int lane = threadIdx.x & 31;
    int row  = blockIdx.x * 8 + (threadIdx.x >> 5);
    if (row >= rows) return;
    float* x = p + (long long)row * cols;
    float mx = -1e30f;
    for (int c = lane; c < cols; c += 32) mx = fmaxf(mx, x[c]);
    mx = wave_max(mx);
    float s = 0.f;
    for (int c = lane; c < cols; c += 32) s += expf(x[c] - mx);
    s = wave_sum(s);
    float inv = 1.f / s;
    for (int c = lane; c < cols; c += 32) x[c] = expf(x[c] - mx) * inv;
}

// out = LayerNorm(X + Y), one wave per row of width D
__global__ void add_ln_kernel(const float* __restrict__ X, const float* __restrict__ Y,
                              const float* __restrict__ g, const float* __restrict__ b,
                              float* __restrict__ out, int rows, int D) {
    int lane = threadIdx.x & 31;
    int row  = blockIdx.x * 8 + (threadIdx.x >> 5);
    if (row >= rows) return;
    const float* xr = X + (long long)row * D;
    const float* yr = Y + (long long)row * D;
    float s = 0.f, sq = 0.f;
    for (int c = lane; c < D; c += 32) { float v = xr[c] + yr[c]; s += v; sq += v * v; }
    s = wave_sum(s); sq = wave_sum(sq);
    float mu  = s / (float)D;
    float var = sq / (float)D - mu * mu;
    float inv = rsqrtf(var + 1e-5f);
    for (int c = lane; c < D; c += 32) {
        float v = xr[c] + yr[c];
        out[(long long)row * D + c] = (v - mu) * inv * g[c] + b[c];
    }
}

__device__ inline float hash01(unsigned x) {
    x ^= x >> 16; x *= 0x7feb352du; x ^= x >> 15; x *= 0x846ca68bu; x ^= x >> 16;
    return (float)(x >> 8) * (1.0f / 16777216.0f);
}

// deterministic inverse-CDF categorical over axis=1 of cm[Bn,Lc,Dc]; sel[b,c]=cm[b,idx,c]
__global__ void star_sample_kernel(const float* __restrict__ cm, float* __restrict__ sel,
                                   int Bn, int Lc, int Dc, unsigned seed) {
    long long t = (long long)blockIdx.x * blockDim.x + threadIdx.x;
    if (t >= (long long)Bn * Dc) return;
    int b = (int)(t / Dc), c = (int)(t % Dc);
    const float* base = cm + (long long)b * Lc * Dc + c;
    float mx = -1e30f;
    for (int l = 0; l < Lc; ++l) mx = fmaxf(mx, base[(long long)l * Dc]);
    float s = 0.f;
    for (int l = 0; l < Lc; ++l) s += expf(base[(long long)l * Dc] - mx);
    float u = hash01(seed * 2654435761u + (unsigned)t) * s;
    float accu = 0.f, val = base[(long long)(Lc - 1) * Dc];
    for (int l = 0; l < Lc; ++l) {
        accu += expf(base[(long long)l * Dc] - mx);
        if (accu >= u) { val = base[(long long)l * Dc]; break; }
    }
    sel[t] = val;
}

// cat[r, 0:D] = x[r, :] ; cat[r, D:D+Dc] = sel[r / Lc, :]
__global__ void concat_kernel(const float* __restrict__ x, const float* __restrict__ sel,
                              float* __restrict__ cat, int rows, int Lc, int D, int Dc) {
    long long idx = (long long)blockIdx.x * blockDim.x + threadIdx.x;
    long long n = (long long)rows * (D + Dc);
    if (idx >= n) return;
    int r = (int)(idx / (D + Dc));
    int c = (int)(idx % (D + Dc));
    int b = r / Lc;
    cat[idx] = (c < D) ? x[(long long)r * D + c] : sel[(long long)b * Dc + (c - D)];
}

// out = a + s + x * w[d] + bb[d]   (depthwise k=1 conv fused with 3-way add)
__global__ void combine_kernel(const float* __restrict__ a, const float* __restrict__ s,
                               const float* __restrict__ x, const float* __restrict__ w,
                               const float* __restrict__ bb, float* __restrict__ out,
                               long long n, int D) {
    long long idx = (long long)blockIdx.x * blockDim.x + threadIdx.x;
    if (idx >= n) return;
    int d = (int)(idx % D);
    out[idx] = a[idx] + s[idx] + x[idx] * w[d] + bb[d];
}

// routers[b, i, :] = outs_i[b, L-1, :]
__global__ void gather_routers_kernel(const float* __restrict__ outs, float* __restrict__ routers) {
    long long idx = (long long)blockIdx.x * blockDim.x + threadIdx.x;
    long long n = (long long)Bsz * NBq * Dm;
    if (idx >= n) return;
    int d = (int)(idx % Dm);
    int i = (int)((idx / Dm) % NBq);
    int b = (int)(idx / ((long long)Dm * NBq));
    routers[idx] = outs[(long long)i * Bsz * Lseq * Dm +
                        ((long long)b * Lseq + (Lseq - 1)) * Dm + d];
}

// out[i,b,l,:] = (l == L-1) ? xinter[b,i,:] : outs_i[b,l,:]
__global__ void scatter_out_kernel(const float* __restrict__ outs, const float* __restrict__ xinter,
                                   float* __restrict__ out) {
    long long idx = (long long)blockIdx.x * blockDim.x + threadIdx.x;
    long long n = (long long)NBq * Bsz * Lseq * Dm;
    if (idx >= n) return;
    int d = (int)(idx % Dm);
    int l = (int)((idx / Dm) % Lseq);
    int b = (int)((idx / ((long long)Dm * Lseq)) % Bsz);
    int i = (int)(idx / ((long long)Dm * Lseq * Bsz));
    if (l == Lseq - 1)
        out[idx] = xinter[((long long)b * NBq + i) * Dm + d];
    else
        out[idx] = outs[(long long)i * Bsz * Lseq * Dm + ((long long)b * Lseq + l) * Dm + d];
}

// ---------------------------------------------------------------------------
// Host-side orchestration
// ---------------------------------------------------------------------------
struct AttnP { const float *Wq,*bq,*Wk,*bk,*Wv,*bv,*Wo,*bo; };
struct EncP  { const float *g1W,*g1b,*g2W,*g2b,*g3W,*g3b,*g4W,*g4b,
               *ln1g,*ln1b,*c1W,*c1b,*c2W,*c2b,*ln2g,*ln2b; };

static inline void gemm(hipStream_t st, const float* A, const float* B, const float* bias,
                        float* C, int M, int N, int K, int lda, int ldb, int ldc,
                        int batch = 1, int bh = 1,
                        long long sA1 = 0, long long sA2 = 0,
                        long long sB1 = 0, long long sB2 = 0,
                        long long sC1 = 0, long long sC2 = 0,
                        int bLayout = 0, int act = 0, float alpha = 1.f)
{
    dim3 grid((N + 127) / 128, (M + 127) / 128, batch);
    wmma_gemm_kernel<<<grid, dim3(256), 0, st>>>(A, B, bias, C, M, N, K, lda, ldb, ldc,
                                                 bh, sA1, sA2, sB1, sB2, sC1, sC2,
                                                 bLayout, act, alpha);
}

static void run_branch(hipStream_t st, const float* x, int Bn, int Ls,
                       const AttnP& ap, const EncP& ep,
                       const float* projW, const float* projb,
                       const float* dwcw, const float* dwcb,
                       float* S0, float* S1, float* S2, float* S3, float* S4,
                       float* BIG, float* sel, float* outdst, unsigned seed)
{
    const int M = Bn * Ls;
    const long long LD = (long long)Ls * Dm;

    // ---- full attention ----
    gemm(st, x, ap.Wq, ap.bq, S0, M, Dm, Dm, Dm, Dm, Dm);
    gemm(st, x, ap.Wk, ap.bk, S1, M, Dm, Dm, Dm, Dm, Dm);
    gemm(st, x, ap.Wv, ap.bv, S2, M, Dm, Dm, Dm, Dm, Dm);
    // scores[b,h,l,s] = q . k^T / sqrt(dh)   (B matrix = K rows, [N=s,K=e] layout)
    gemm(st, S0, S1, nullptr, BIG, Ls, Ls, Dh, Dm, Dm, Ls,
         Bn * Hh, Hh, LD, Dh, LD, Dh,
         (long long)Hh * Ls * Ls, (long long)Ls * Ls, 1, 0, 0.125f);
    {
        int rows = Bn * Hh * Ls;
        softmax_rows_kernel<<<(rows + 7) / 8, 256, 0, st>>>(BIG, rows, Ls);
    }
    // o[b,l,h,d] = A . V
    gemm(st, BIG, S2, nullptr, S3, Ls, Dh, Ls, Ls, Dm, Dm,
         Bn * Hh, Hh, (long long)Hh * Ls * Ls, (long long)Ls * Ls,
         LD, Dh, LD, Dh, 0, 0, 1.f);
    gemm(st, S3, ap.Wo, ap.bo, S4, M, Dm, Dm, Dm, Dm, Dm);      // a = S4

    // ---- STAR ----
    gemm(st, x,  ep.g1W, ep.g1b, S0, M, Dm,  Dm, Dm, Dm,  Dm, 1,1,0,0,0,0,0,0, 0, 2, 1.f);
    gemm(st, S0, ep.g2W, ep.g2b, S1, M, DCc, Dm, Dm, DCc, DCc);
    {
        long long n = (long long)Bn * DCc;
        star_sample_kernel<<<(int)((n + 255) / 256), 256, 0, st>>>(S1, sel, Bn, Ls, DCc, seed);
        long long nc = (long long)M * (Dm + DCc);
        concat_kernel<<<(int)((nc + 255) / 256), 256, 0, st>>>(x, sel, BIG, M, Ls, Dm, DCc);
    }
    gemm(st, BIG, ep.g3W, ep.g3b, S2, M, Dm, Dm + DCc, Dm + DCc, Dm, Dm,
         1,1,0,0,0,0,0,0, 0, 2, 1.f);
    gemm(st, S2, ep.g4W, ep.g4b, S0, M, Dm, Dm, Dm, Dm, Dm);     // star = S0
    add_ln_kernel<<<(M + 7) / 8, 256, 0, st>>>(x, S0, ep.ln1g, ep.ln1b, S1, M, Dm); // x1 = S1

    // ---- FFN ----
    gemm(st, S1, ep.c1W, ep.c1b, BIG, M, DFFc, Dm, Dm, DFFc, DFFc,
         1,1,0,0,0,0,0,0, 0, 1, 1.f);
    gemm(st, BIG, ep.c2W, ep.c2b, S2, M, Dm, DFFc, DFFc, Dm, Dm);
    add_ln_kernel<<<(M + 7) / 8, 256, 0, st>>>(S1, S2, ep.ln2g, ep.ln2b, S3, M, Dm); // s = S3

    // ---- combine + proj ----
    {
        long long n = (long long)M * Dm;
        combine_kernel<<<(int)((n + 255) / 256), 256, 0, st>>>(S4, S3, x, dwcw, dwcb, S0, n, Dm);
    }
    gemm(st, S0, projW, projb, outdst, M, Dm, Dm, Dm, Dm, Dm);
}

extern "C" void kernel_launch(void* const* d_in, const int* in_sizes, int n_in,
                              void* d_out, int out_size, void* d_ws, size_t ws_size,
                              hipStream_t stream)
{
    (void)in_sizes; (void)n_in; (void)out_size; (void)ws_size;
    const float* x = (const float*)d_in[0];

    // intra attn params (each has leading NB dim)
    const float* iaW[8]; for (int j = 0; j < 8; ++j) iaW[j] = (const float*)d_in[1 + j];
    // intra enc params (16, leading NB dim)
    const float* ieW[16]; for (int j = 0; j < 16; ++j) ieW[j] = (const float*)d_in[9 + j];
    // inter attn (8) + inter enc (16)
    const float* xaW[8]; for (int j = 0; j < 8; ++j) xaW[j] = (const float*)d_in[25 + j];
    const float* xeW[16]; for (int j = 0; j < 16; ++j) xeW[j] = (const float*)d_in[33 + j];
    // shared
    const float* projW = (const float*)d_in[49];
    const float* projb = (const float*)d_in[50];
    const float* dwcw  = (const float*)d_in[51];
    const float* dwcb  = (const float*)d_in[52];

    // workspace layout (floats)
    float* ws = (float*)d_ws;
    const long long BLD = (long long)Bsz * Lseq * Dm;               // 4M floats
    float* S0  = ws;
    float* S1  = S0 + BLD;
    float* S2  = S1 + BLD;
    float* S3  = S2 + BLD;
    float* S4  = S3 + BLD;
    float* BIG = S4 + BLD;                                          // 33.5M floats
    float* OUTS = BIG + (long long)Bsz * Hh * Lseq * Lseq;          // 4 x BLD
    float* sel     = OUTS + 4 * BLD;
    float* routers = sel + (long long)Bsz * DCc;
    float* xinter  = routers + (long long)Bsz * NBq * Dm;

    // per-block intra parameter strides
    const long long sDD = (long long)Dm * Dm, sD = Dm;
    const long long sDDC = (long long)Dm * DCc, sDC = DCc;
    const long long sCAT = (long long)(Dm + DCc) * Dm;
    const long long sDF = (long long)Dm * DFFc, sF = DFFc;
    const long long sFD = (long long)DFFc * Dm;

    for (int i = 0; i < NBq; ++i) {
        AttnP ap = { iaW[0] + i*sDD, iaW[1] + i*sD, iaW[2] + i*sDD, iaW[3] + i*sD,
                     iaW[4] + i*sDD, iaW[5] + i*sD, iaW[6] + i*sDD, iaW[7] + i*sD };
        EncP ep = { ieW[0] + i*sDD,  ieW[1] + i*sD,  ieW[2] + i*sDDC, ieW[3] + i*sDC,
                    ieW[4] + i*sCAT, ieW[5] + i*sD,  ieW[6] + i*sDD,  ieW[7] + i*sD,
                    ieW[8] + i*sD,   ieW[9] + i*sD,  ieW[10] + i*sDF, ieW[11] + i*sF,
                    ieW[12] + i*sFD, ieW[13] + i*sD, ieW[14] + i*sD,  ieW[15] + i*sD };
        run_branch(stream, x + i * BLD, Bsz, Lseq, ap, ep,
                   projW, projb, dwcw, dwcb,
                   S0, S1, S2, S3, S4, BIG, sel,
                   OUTS + i * BLD, (unsigned)(0x9e3779b9u + 17u * i));
    }

    {
        long long n = (long long)Bsz * NBq * Dm;
        gather_routers_kernel<<<(int)((n + 255) / 256), 256, 0, stream>>>(OUTS, routers);
    }

    AttnP ap2 = { xaW[0], xaW[1], xaW[2], xaW[3], xaW[4], xaW[5], xaW[6], xaW[7] };
    EncP  ep2 = { xeW[0], xeW[1], xeW[2], xeW[3], xeW[4], xeW[5], xeW[6], xeW[7],
                  xeW[8], xeW[9], xeW[10], xeW[11], xeW[12], xeW[13], xeW[14], xeW[15] };
    run_branch(stream, routers, Bsz, NBq, ap2, ep2,
               projW, projb, dwcw, dwcb,
               S0, S1, S2, S3, S4, BIG, sel, xinter, 0xC0FFEEu);

    {
        long long n = (long long)NBq * Bsz * Lseq * Dm;
        scatter_out_kernel<<<(int)((n + 255) / 256), 256, 0, stream>>>(OUTS, xinter, (float*)d_out);
    }
}
